// TextEncoderBlock_78056735637552
// MI455X (gfx1250) — compile-verified, matrix operand order checked by reading, not run on hardware
//
#include <hip/hip_runtime.h>
#include <cstdint>
#include <cstddef>

// ---------------------------------------------------------------------------
// Problem constants (from reference)
// ---------------------------------------------------------------------------
constexpr int Bc   = 2;
constexpr int Lc   = 2048;
constexpr int Ec   = 2048;
constexpr int QHc  = 16;
constexpr int KVHc = 4;
constexpr int Dc   = 128;
constexpr int FFNc = 8192;
constexpr int BLc  = Bc * Lc;          // 4096 tokens

typedef __bf16 bf16_t;
typedef __attribute__((ext_vector_type(8)))  __bf16 v8bf;
typedef __attribute__((ext_vector_type(16))) __bf16 v16bf;
typedef __attribute__((ext_vector_type(8)))  float  v8f;

// ---------------------------------------------------------------------------
// gfx1250 async global->LDS copy (ASYNCcnt-tracked), with safe fallback.
// Builtin signature (from compiler diagnostic): param0 = v4i in AS1 (global),
// param1 = v4i in AS3 (LDS), then two int immediates (offset, cpol).
// ---------------------------------------------------------------------------
#if defined(__has_builtin)
#  if __has_builtin(__builtin_amdgcn_global_load_async_to_lds_b128)
#    define HAVE_ASYNC_LDS 1
#  endif
#  if __has_builtin(__builtin_amdgcn_s_wait_asynccnt)
#    define HAVE_WAIT_ASYNC 1
#  endif
#endif
#ifndef HAVE_ASYNC_LDS
#  define HAVE_ASYNC_LDS 0
#endif
#ifndef HAVE_WAIT_ASYNC
#  define HAVE_WAIT_ASYNC 0
#endif

typedef __attribute__((__vector_size__(4 * sizeof(int)))) int v4i;
typedef __attribute__((address_space(1))) v4i gv4i;   // global int4
typedef __attribute__((address_space(3))) v4i sv4i;   // LDS int4

__device__ __forceinline__ void async_copy16(const bf16_t* g, bf16_t* l) {
#if HAVE_ASYNC_LDS
  __builtin_amdgcn_global_load_async_to_lds_b128(
      (gv4i*)(uintptr_t)g,
      (sv4i*)(uint32_t)(uintptr_t)l,
      0, 0);
#else
  *(v8bf*)l = *(const v8bf*)g;
#endif
}

template <int N>
__device__ __forceinline__ void wait_async_tile() {
#if HAVE_ASYNC_LDS
#  if HAVE_WAIT_ASYNC
  __builtin_amdgcn_s_wait_asynccnt((unsigned short)N);
#  else
  asm volatile("s_wait_asynccnt %0" ::"n"(N) : "memory");
#  endif
#endif
}

// ---------------------------------------------------------------------------
// WMMA helpers (CDNA5: V_WMMA_F32_16X16X32_BF16, wave32)
// A fragment: lane m = lane&15, K halves {kb..kb+7} and {kb+16..kb+23},
// kb = (lane>>4)*8.  B fragment is symmetric with the tile stored
// outer-dim-major (so both use the same loader).
// ---------------------------------------------------------------------------
__device__ __forceinline__ v16bf load_frag(const bf16_t* base, int stride,
                                           int row0, int kofs, int lane) {
  const int r  = lane & 15;
  const int kb = kofs + ((lane >> 4) << 3);
  const bf16_t* p = base + (size_t)(row0 + r) * stride + kb;
  v8bf lo = *(const v8bf*)p;
  v8bf hi = *(const v8bf*)(p + 16);
  v16bf out;
#pragma unroll
  for (int i = 0; i < 8; ++i) { out[i] = lo[i]; out[i + 8] = hi[i]; }
  return out;
}

__device__ __forceinline__ v8f wmma_bf16(v16bf a, v16bf b, v8f c) {
  return __builtin_amdgcn_wmma_f32_16x16x32_bf16(
      /*neg_a=*/false, a, /*neg_b=*/false, b,
      /*c_mod=*/(short)0, c, /*reuse_a=*/false, /*reuse_b=*/false);
}

// ---------------------------------------------------------------------------
// Weight transpose + fp32->bf16: Wt[N][K] = (bf16) W[K][N]
// 32x32 LDS tile, coalesced read (along N) and write (along K).
// ---------------------------------------------------------------------------
__global__ __launch_bounds__(256) void transpose_convert_kernel(
    const float* __restrict__ W, bf16_t* __restrict__ Wt, int K, int N) {
  __shared__ bf16_t t[32][33];
  const int k0 = blockIdx.y * 32;
  const int n0 = blockIdx.x * 32;
  const int tx = threadIdx.x & 31;
  const int ty = threadIdx.x >> 5;   // 0..7
#pragma unroll
  for (int i = 0; i < 4; ++i)
    t[ty + i * 8][tx] = (bf16_t)W[(size_t)(k0 + ty + i * 8) * N + (n0 + tx)];
  __syncthreads();
#pragma unroll
  for (int i = 0; i < 4; ++i)
    Wt[(size_t)(n0 + ty + i * 8) * K + (k0 + tx)] = t[tx][ty + i * 8];
}

// ---------------------------------------------------------------------------
// RMSNorm: one block per token row of E floats, bf16 output
// ---------------------------------------------------------------------------
__global__ __launch_bounds__(256) void rmsnorm_kernel(
    const float* __restrict__ x, const float* __restrict__ w,
    bf16_t* __restrict__ out) {
  const int row = blockIdx.x;
  const float* xr = x + (size_t)row * Ec;
  __shared__ float red[256];
  float s = 0.f;
  for (int j = threadIdx.x; j < Ec; j += 256) { float v = xr[j]; s += v * v; }
  red[threadIdx.x] = s;
  __syncthreads();
  for (int off = 128; off > 0; off >>= 1) {
    if (threadIdx.x < off) red[threadIdx.x] += red[threadIdx.x + off];
    __syncthreads();
  }
  const float r = rsqrtf(red[0] / (float)Ec + 1.1920929e-7f);
  bf16_t* o = out + (size_t)row * Ec;
  for (int j = threadIdx.x; j < Ec; j += 256) o[j] = (bf16_t)(xr[j] * r * w[j]);
}

// ---------------------------------------------------------------------------
// GEMM: C[M,N] (f32, + optional fp32 residual) = A[M,K] bf16 x Wt[N,K] bf16
// 128x64 block tile, BK=64, 8 waves, 4 accumulators/wave (32 WMMA sub-tiles).
// Double-buffered LDS, async global->LDS copies pipelined against WMMA.
// M multiple of 128, N multiple of 64, K multiple of 64 (true for all calls).
// ---------------------------------------------------------------------------
#define GBM 128
#define GBN 64
#define GBK 64
#define GLS 72   // padded LDS row stride (halves); 144B, 16B-aligned chunks

__global__ __launch_bounds__(256) void gemm_bf16_kernel(
    const bf16_t* __restrict__ A, const bf16_t* __restrict__ Wt,
    const float* __restrict__ res, float* __restrict__ C,
    int M, int N, int K) {
  __shared__ bf16_t As[2][GBM][GLS];   // [m][k]
  __shared__ bf16_t Bs[2][GBN][GLS];   // [n][k]

  const int tid  = threadIdx.x;
  const int lane = tid & 31;
  const int w    = tid >> 5;
  const int bm   = blockIdx.y * GBM;
  const int bn   = blockIdx.x * GBN;

  const int ni  = w & 3;            // 0..3  (16-col strip)
  const int mi0 = (w >> 2) * 4;     // 0 or 4 (four 16-row strips)

  v8f acc[4];
#pragma unroll
  for (int j = 0; j < 4; ++j)
    acc[j] = (v8f){0.f, 0.f, 0.f, 0.f, 0.f, 0.f, 0.f, 0.f};

  // 6 chunk-copies per thread per tile (A: 4, B: 2)
  auto issue_tile = [&](int k0, int buf) {
#pragma unroll
    for (int i = 0; i < 4; ++i) {
      const int c = tid + 256 * i;
      const int r = c >> 3, ch = (c & 7) * 8;
      async_copy16(A + (size_t)(bm + r) * K + k0 + ch, &As[buf][r][ch]);
    }
#pragma unroll
    for (int i = 0; i < 2; ++i) {
      const int c = tid + 256 * i;
      const int r = c >> 3, ch = (c & 7) * 8;
      async_copy16(Wt + (size_t)(bn + r) * K + k0 + ch, &Bs[buf][r][ch]);
    }
  };

  const int nt = K / GBK;
  issue_tile(0, 0);

  for (int i = 0; i < nt; ++i) {
    if (i + 1 < nt) {
      issue_tile((i + 1) * GBK, (i + 1) & 1);  // prefetch next tile
      wait_async_tile<6>();                    // current tile's 6 copies done
    } else {
      wait_async_tile<0>();
    }
    __syncthreads();

    const bf16_t* Asc = &As[i & 1][0][0];
    const bf16_t* Bsc = &Bs[i & 1][0][0];
#pragma unroll
    for (int kk = 0; kk < GBK; kk += 32) {
      v16bf bb = load_frag(Bsc, GLS, ni * 16, kk, lane);
#pragma unroll
      for (int j = 0; j < 4; ++j) {
        v16bf aa = load_frag(Asc, GLS, (mi0 + j) * 16, kk, lane);
        acc[j] = wmma_bf16(aa, bb, acc[j]);
      }
    }
    __syncthreads();
  }

  // C layout: lane -> col = lane&15, row offset 8*(lane>>4), vgpr r -> row+r
  const int col  = lane & 15;
  const int rofs = 8 * (lane >> 4);
  const int cc   = bn + ni * 16 + col;
#pragma unroll
  for (int j = 0; j < 4; ++j) {
#pragma unroll
    for (int r = 0; r < 8; ++r) {
      size_t idx = (size_t)(bm + (mi0 + j) * 16 + rofs + r) * N + cc;
      float v = acc[j][r];
      if (res) v += res[idx];
      C[idx] = v;
    }
  }
}

// ---------------------------------------------------------------------------
// RoPE on q,k (+ bf16 convert). V is written TRANSPOSED: vb[b][kvh][d][L]
// so the attention kernel can fill its Vt LDS tile with b128 copies.
// One thread per (token, head, i<D/2).
// ---------------------------------------------------------------------------
__global__ __launch_bounds__(256) void rope_qkv_kernel(
    const float* __restrict__ qf, const float* __restrict__ kf,
    const float* __restrict__ vf, const float* __restrict__ ts,
    bf16_t* __restrict__ qb, bf16_t* __restrict__ kb, bf16_t* __restrict__ vb) {
  const int idx = blockIdx.x * blockDim.x + threadIdx.x;   // BL*QH*64
  const int i = idx & 63;
  const int h = (idx >> 6) & (QHc - 1);
  const int t = idx >> 10;                                 // token 0..BL-1
  const int pos = t & (Lc - 1);
  const int bidx = t >> 11;                                // t / Lc
  const float rad = (float)pos / ts[i];
  const float s = __sinf(rad), c = __cosf(rad);

  {
    const float* qp = qf + ((size_t)t * QHc + h) * Dc;
    bf16_t* qo = qb + ((size_t)t * QHc + h) * Dc;
    const float x1 = qp[i], x2 = qp[i + 64];
    qo[i]      = (bf16_t)(x1 * c - x2 * s);
    qo[i + 64] = (bf16_t)(x2 * c + x1 * s);
  }
  if (h < KVHc) {
    const float* kp = kf + ((size_t)t * KVHc + h) * Dc;
    bf16_t* ko = kb + ((size_t)t * KVHc + h) * Dc;
    const float y1 = kp[i], y2 = kp[i + 64];
    ko[i]      = (bf16_t)(y1 * c - y2 * s);
    ko[i + 64] = (bf16_t)(y2 * c + y1 * s);
    const float* vp = vf + ((size_t)t * KVHc + h) * Dc;
    bf16_t* vo = vb + (((size_t)bidx * KVHc + h) * Dc) * Lc;  // [d][L]
    vo[(size_t)i * Lc + pos]        = (bf16_t)vp[i];
    vo[(size_t)(i + 64) * Lc + pos] = (bf16_t)vp[i + 64];
  }
}

// ---------------------------------------------------------------------------
// Flash attention (causal, GQA 16q/4kv). Block = (query tile of 64, head, b).
// S = Q K^T and O += P V through WMMA; online softmax in LDS; PV accumulators
// held in registers (mi = w>>1, 4 column sub-tiles per wave).
// ---------------------------------------------------------------------------
#define QKS 136   // Q/K LDS row stride (halves)  -> 272B, 16B-aligned
#define VTS 72    // Vt / P LDS row stride (halves) -> 144B, 16B-aligned

__global__ __launch_bounds__(256) void attention_kernel(
    const bf16_t* __restrict__ qb, const bf16_t* __restrict__ kb,
    const bf16_t* __restrict__ vt, bf16_t* __restrict__ ctx) {
  extern __shared__ __align__(16) char smem[];
  bf16_t* Qs   = (bf16_t*)smem;                  // [64][QKS]
  bf16_t* Ks   = Qs + 64 * QKS;                  // [64][QKS]
  bf16_t* Vt   = Ks + 64 * QKS;                  // [128][VTS] (d-major)
  float*  Sb   = (float*)(Vt + 128 * VTS);       // [64][64]
  bf16_t* Pb   = (bf16_t*)(Sb + 64 * 64);        // [64][VTS]
  float*  mrow = (float*)(Pb + 64 * VTS);        // [64]
  float*  lrow = mrow + 64;                      // [64]
  float*  arow = lrow + 64;                      // [64]

  const int tid  = threadIdx.x;
  const int lane = tid & 31;
  const int w    = tid >> 5;
  const int qblk = blockIdx.x;       // 0..31
  const int h    = blockIdx.y;       // 0..15
  const int b    = blockIdx.z;       // 0..1
  const int kvh  = h >> 2;           // G = 4
  const int q0   = qblk * 64;

  // load Q tile (64 x 128 halves) -- async; fenced by first loop iteration
  for (int cch = tid; cch < 64 * 16; cch += 256) {
    const int r  = cch >> 4;
    const int ch = (cch & 15) * 8;
    async_copy16(qb + (((size_t)(b * Lc + q0 + r) * QHc + h) * Dc) + ch,
                 &Qs[r * QKS + ch]);
  }
  if (tid < 64) { mrow[tid] = -3.0e38f; lrow[tid] = 0.f; }

  const int mi   = w >> 1;           // 0..3  (16-row strip)
  const int nset = (w & 1) * 4;      // 0 or 4 (four 16-col d sub-tiles)
  v8f oacc[4];
#pragma unroll
  for (int j = 0; j < 4; ++j)
    oacc[j] = (v8f){0.f, 0.f, 0.f, 0.f, 0.f, 0.f, 0.f, 0.f};

  const bf16_t* vbase = vt + (((size_t)b * KVHc + kvh) * Dc) * Lc;

  for (int kbk = 0; kbk <= qblk; ++kbk) {
    const int k0 = kbk * 64;

    // K tile row-major, V tile d-major (async b128 copies)
    for (int cch = tid; cch < 64 * 16; cch += 256) {
      const int r  = cch >> 4;
      const int ch = (cch & 15) * 8;
      const size_t gro = ((size_t)(b * Lc + k0 + r) * KVHc + kvh) * Dc;
      async_copy16(kb + gro + ch, &Ks[r * QKS + ch]);
    }
    for (int cch = tid; cch < 128 * 8; cch += 256) {
      const int d  = cch >> 3;
      const int ch = (cch & 7) * 8;
      async_copy16(vbase + (size_t)d * Lc + k0 + ch, &Vt[d * VTS + ch]);
    }
    wait_async_tile<0>();
    __syncthreads();

    // ---- S = Q K^T  (16 sub-tiles over 8 waves, K-dim = D = 128) ----
    {
      const int sni  = w & 3;
      const int smi0 = (w >> 2) * 2;
      v8f s0 = {0.f, 0.f, 0.f, 0.f, 0.f, 0.f, 0.f, 0.f};
      v8f s1 = {0.f, 0.f, 0.f, 0.f, 0.f, 0.f, 0.f, 0.f};
#pragma unroll
      for (int ks = 0; ks < 4; ++ks) {
        v16bf bbf = load_frag(Ks, QKS, sni * 16,       ks * 32, lane);
        v16bf a0  = load_frag(Qs, QKS, smi0 * 16,      ks * 32, lane);
        v16bf a1  = load_frag(Qs, QKS, smi0 * 16 + 16, ks * 32, lane);
        s0 = wmma_bf16(a0, bbf, s0);
        s1 = wmma_bf16(a1, bbf, s1);
      }
      const float scale = 0.08838834764831845f;   // 1/sqrt(128)
      const int col  = lane & 15;
      const int rofs = 8 * (lane >> 4);
#pragma unroll
      for (int r = 0; r < 8; ++r) {
        Sb[(smi0 * 16 + rofs + r) * 64 + sni * 16 + col]      = s0[r] * scale;
        Sb[(smi0 * 16 + 16 + rofs + r) * 64 + sni * 16 + col] = s1[r] * scale;
      }
    }
    __syncthreads();

    // ---- online softmax over the 64x64 score tile ----
    if (tid < 64) {
      const int r = tid;
      const int nvalid = (kbk == qblk) ? (r + 1) : 64;   // causal diagonal
      const float mold = mrow[r];
      float mx = mold;
      for (int c2 = 0; c2 < nvalid; ++c2) mx = fmaxf(mx, Sb[r * 64 + c2]);
      const float alpha = __expf(mold - mx);
      float l = lrow[r] * alpha;
      for (int c2 = 0; c2 < 64; ++c2) {
        float p = 0.f;
        if (c2 < nvalid) { p = __expf(Sb[r * 64 + c2] - mx); l += p; }
        Pb[r * VTS + c2] = (bf16_t)p;
      }
      mrow[r] = mx; lrow[r] = l; arow[r] = alpha;
    }
    __syncthreads();

    // ---- rescale O and O += P @ V ----
    {
      const int rofs = 8 * (lane >> 4);
      float al[8];
#pragma unroll
      for (int r = 0; r < 8; ++r) al[r] = arow[mi * 16 + rofs + r];
#pragma unroll
      for (int j = 0; j < 4; ++j)
#pragma unroll
        for (int r = 0; r < 8; ++r) oacc[j][r] *= al[r];

#pragma unroll
      for (int ks = 0; ks < 2; ++ks) {
        v16bf a = load_frag(Pb, VTS, mi * 16, ks * 32, lane);
#pragma unroll
        for (int j = 0; j < 4; ++j) {
          v16bf bbf = load_frag(Vt, VTS, (nset + j) * 16, ks * 32, lane);
          oacc[j] = wmma_bf16(a, bbf, oacc[j]);
        }
      }
    }
    __syncthreads();
  }

  // ---- epilogue: normalize and store ctx (bf16) ----
  {
    const int col  = lane & 15;
    const int rofs = 8 * (lane >> 4);
#pragma unroll
    for (int r = 0; r < 8; ++r) {
      const int row = mi * 16 + rofs + r;
      const float invl = 1.f / lrow[row];
      const size_t base = ((size_t)(b * Lc + q0 + row) * QHc + h) * Dc;
#pragma unroll
      for (int j = 0; j < 4; ++j)
        ctx[base + (nset + j) * 16 + col] = (bf16_t)(oacc[j][r] * invl);
    }
  }
}

// ---------------------------------------------------------------------------
// SwiGLU combine: t = bf16( silu(g) * u )
// ---------------------------------------------------------------------------
__global__ __launch_bounds__(256) void silu_mul_kernel(
    const float* __restrict__ g, const float* __restrict__ u,
    bf16_t* __restrict__ t, size_t n) {
  size_t i = (size_t)blockIdx.x * blockDim.x + threadIdx.x;
  size_t stride = (size_t)gridDim.x * blockDim.x;
  for (; i < n; i += stride) {
    const float gv = g[i];
    const float sg = gv / (1.f + __expf(-gv));
    t[i] = (bf16_t)(sg * u[i]);
  }
}

// ---------------------------------------------------------------------------
// Host orchestration
// ---------------------------------------------------------------------------
static inline void launch_gemm(const bf16_t* A, const bf16_t* Wt,
                               const float* res, float* C,
                               int M, int N, int K, hipStream_t s) {
  dim3 grid(N / GBN, M / GBM);
  gemm_bf16_kernel<<<grid, 256, 0, s>>>(A, Wt, res, C, M, N, K);
}

extern "C" void kernel_launch(void* const* d_in, const int* in_sizes, int n_in,
                              void* d_out, int out_size, void* d_ws, size_t ws_size,
                              hipStream_t stream) {
  (void)in_sizes; (void)n_in; (void)out_size; (void)ws_size;

  const float* x_in   = (const float*)d_in[0];
  const float* ln1_w  = (const float*)d_in[1];
  const float* wq     = (const float*)d_in[2];
  const float* wk     = (const float*)d_in[3];
  const float* wv     = (const float*)d_in[4];
  const float* wo     = (const float*)d_in[5];
  const float* ln2_w  = (const float*)d_in[6];
  const float* wg     = (const float*)d_in[7];
  const float* wu     = (const float*)d_in[8];
  const float* wd     = (const float*)d_in[9];
  const float* ropets = (const float*)d_in[10];

  float* x_out = (float*)d_out;                                  // B*L*E
  float* k_out = x_out + (size_t)BLc * Ec;                       // B*L*KVH*D
  float* v_out = k_out + (size_t)BLc * KVHc * Dc;

  // ---- workspace carve ----
  char* ws = (char*)d_ws;
  size_t off = 0;
  auto carve = [&](size_t bytes) -> char* {
    char* p = ws + off;
    off = (off + bytes + 255) & ~(size_t)255;
    return p;
  };
  const size_t QKV_N = (size_t)QHc * Dc;       // 2048
  const size_t KV_N  = (size_t)KVHc * Dc;      // 512

  bf16_t* wq_t = (bf16_t*)carve((size_t)Ec * QKV_N * 2);   // [N][K]
  bf16_t* wk_t = (bf16_t*)carve((size_t)Ec * KV_N * 2);
  bf16_t* wv_t = (bf16_t*)carve((size_t)Ec * KV_N * 2);
  bf16_t* wo_t = (bf16_t*)carve((size_t)QKV_N * Ec * 2);
  bf16_t* wg_t = (bf16_t*)carve((size_t)Ec * FFNc * 2);
  bf16_t* wu_t = (bf16_t*)carve((size_t)Ec * FFNc * 2);
  bf16_t* wd_t = (bf16_t*)carve((size_t)FFNc * Ec * 2);
  bf16_t* h_b  = (bf16_t*)carve((size_t)BLc * Ec * 2);
  float*  x1   = (float*)carve((size_t)BLc * Ec * 4);

  // Attention-phase region; aliased by FFN-phase buffers afterwards.
  const size_t phase_off = off;
  float*  q_f   = (float*)carve((size_t)BLc * QKV_N * 4);
  float*  k_f   = (float*)carve((size_t)BLc * KV_N * 4);
  float*  v_f   = (float*)carve((size_t)BLc * KV_N * 4);
  bf16_t* q_b   = (bf16_t*)carve((size_t)BLc * QKV_N * 2);
  bf16_t* k_b   = (bf16_t*)carve((size_t)BLc * KV_N * 2);
  bf16_t* v_t   = (bf16_t*)carve((size_t)BLc * KV_N * 2);   // [b][kvh][d][L]
  bf16_t* ctx_b = (bf16_t*)carve((size_t)BLc * QKV_N * 2);

  off = phase_off;  // FFN phase reuses the attention region
  float*  g_f = (float*)carve((size_t)BLc * FFNc * 4);
  float*  u_f = (float*)carve((size_t)BLc * FFNc * 4);
  bf16_t* t_b = (bf16_t*)carve((size_t)BLc * FFNc * 2);

  // ---- 1) weights fp32 -> bf16, transposed to [N][K] ----
  auto tcvt = [&](const float* src, bf16_t* dst, int K, int N) {
    dim3 grid(N / 32, K / 32);
    transpose_convert_kernel<<<grid, 256, 0, stream>>>(src, dst, K, N);
  };
  tcvt(wq, wq_t, Ec, (int)QKV_N);
  tcvt(wk, wk_t, Ec, (int)KV_N);
  tcvt(wv, wv_t, Ec, (int)KV_N);
  tcvt(wo, wo_t, (int)QKV_N, Ec);
  tcvt(wg, wg_t, Ec, FFNc);
  tcvt(wu, wu_t, Ec, FFNc);
  tcvt(wd, wd_t, FFNc, Ec);

  // ---- 2) h = rmsnorm(x, ln1) ----
  rmsnorm_kernel<<<BLc, 256, 0, stream>>>(x_in, ln1_w, h_b);

  // ---- 3) QKV projections ----
  launch_gemm(h_b, wq_t, nullptr, q_f, BLc, (int)QKV_N, Ec, stream);
  launch_gemm(h_b, wk_t, nullptr, k_f, BLc, (int)KV_N, Ec, stream);
  launch_gemm(h_b, wv_t, nullptr, v_f, BLc, (int)KV_N, Ec, stream);

  // ---- 4) RoPE + bf16 pack (V transposed to d-major) ----
  rope_qkv_kernel<<<(BLc * QHc * (Dc / 2)) / 256, 256, 0, stream>>>(
      q_f, k_f, v_f, ropets, q_b, k_b, v_t);

  // ---- 5) flash attention ----
  {
    const size_t att_smem =
        (size_t)(64 * QKS + 64 * QKS + 128 * VTS) * 2 +   // Qs, Ks, Vt
        (size_t)64 * 64 * 4 +                             // Sb
        (size_t)64 * VTS * 2 +                            // Pb
        (size_t)64 * 4 * 3;                               // mrow/lrow/arow
    dim3 grid(Lc / 64, QHc, Bc);
    attention_kernel<<<grid, 256, att_smem, stream>>>(q_b, k_b, v_t, ctx_b);
  }

  // ---- 6) x1 = ctx @ wo + x ----
  launch_gemm(ctx_b, wo_t, x_in, x1, BLc, Ec, (int)QKV_N, stream);

  // ---- 7) h = rmsnorm(x1, ln2) ----
  rmsnorm_kernel<<<BLc, 256, 0, stream>>>(x1, ln2_w, h_b);

  // ---- 8) FFN: g, u, swiglu, down + residual ----
  launch_gemm(h_b, wg_t, nullptr, g_f, BLc, FFNc, Ec, stream);
  launch_gemm(h_b, wu_t, nullptr, u_f, BLc, FFNc, Ec, stream);
  silu_mul_kernel<<<8192, 256, 0, stream>>>(g_f, u_f, t_b, (size_t)BLc * FFNc);
  launch_gemm(t_b, wd_t, x1, x_out, BLc, Ec, FFNc, stream);

  // ---- 9) outputs: h_out = rmsnorm(x_out, ln1); k_out/v_out projections ----
  rmsnorm_kernel<<<BLc, 256, 0, stream>>>(x_out, ln1_w, h_b);
  launch_gemm(h_b, wk_t, nullptr, k_out, BLc, (int)KV_N, Ec, stream);
  launch_gemm(h_b, wv_t, nullptr, v_out, BLc, (int)KV_N, Ec, stream);
}
